// FlashSVDFFN_85529978733276
// MI455X (gfx1250) — matrix operand order repaired and checked by reading, not compiled, not run on hardware
//
#include <hip/hip_runtime.h>
#include <math.h>

#define D_MODEL 1024
#define D_FF    4096
#define RANK    256
#define NROWS   16384   // B*S = 4*4096
#define M_TILE  64
#define FC      64

typedef __attribute__((ext_vector_type(16))) __bf16        v16bf;
typedef __attribute__((ext_vector_type(8)))  float         v8f;
typedef __attribute__((ext_vector_type(4)))  unsigned int  uint4v;
typedef __attribute__((ext_vector_type(4)))  float         float4v;
typedef __attribute__((ext_vector_type(4)))  unsigned int  u32x4;
typedef __attribute__((ext_vector_type(8)))  int           i32x8;
typedef __attribute__((ext_vector_type(4)))  int           i32x4;

union Frag { v16bf v; uint4v q[2]; };

// Native f32 -> bf16 (compiler emits hardware convert, RNE)
__device__ __forceinline__ unsigned short f2bf(float f) {
    union { __bf16 h; unsigned short u; } c;
    c.h = (__bf16)f;
    return c.u;
}

__device__ __forceinline__ float gelu_tanh(float v) {
    float u = 0.7978845608028654f * (v + 0.044715f * v * v * v);
    return 0.5f * v * (1.0f + tanhf(u));
}

#define WMMA_BF16(a, b, c) \
    __builtin_amdgcn_wmma_f32_16x16x32_bf16(false, (a), false, (b), (short)0, (c), false, false)

// ---------------------------------------------------------------------------
// Tensor Data Mover: async 2D tile load global -> LDS (one instruction).
// D# per CDNA5 ISA. data_size = 1 byte. Groups 2/3 = 0
// (2D tile; tile_dim2 = 0 so higher dims unused). Issued by one wave; caller
// waits on TENSORcnt before the workgroup barrier.
// ---------------------------------------------------------------------------
__device__ __forceinline__ void tdm_load_2d(unsigned lds_byte,       // LDS dest (byte addr)
                                            const void* gsrc,        // global src
                                            unsigned tile_bytes,     // bytes per row
                                            unsigned tile_rows,      // #rows
                                            unsigned row_stride) {   // row stride (bytes)
    unsigned long long ga = (unsigned long long)(size_t)gsrc;
    u32x4 g0;
    g0.x = 1u;                                                  // count=1 (valid, user)
    g0.y = lds_byte;                                            // lds_addr
    g0.z = (unsigned)ga;                                        // global_addr[31:0]
    g0.w = ((unsigned)(ga >> 32) & 0x01FFFFFFu) | (2u << 30);   // addr[56:32] | type=2
    i32x8 g1;
    g1[0] = 0;                                                  // wg_mask=0, data_size=1B
    g1[1] = (int)((row_stride & 0xFFFFu) << 16);                // tensor_dim0 lo16
    g1[2] = (int)((row_stride >> 16) | ((tile_rows & 0xFFFFu) << 16)); // dim0 hi | dim1 lo
    g1[3] = (int)(tile_bytes << 16);                            // dim1 hi=0 | tile_dim0
    g1[4] = (int)(tile_rows & 0xFFFFu);                         // tile_dim1 | tile_dim2=0
    g1[5] = (int)row_stride;                                    // tensor_dim0_stride lo32
    g1[6] = 0;                                                  // stride hi | dim1_stride lo
    g1[7] = 0;
    i32x4 z = {0, 0, 0, 0};
#if defined(__clang_major__) && __clang_major__ >= 23
    i32x8 z8 = {0, 0, 0, 0, 0, 0, 0, 0};
    __builtin_amdgcn_tensor_load_to_lds(g0, g1, z, z, z8, 0);
#else
    __builtin_amdgcn_tensor_load_to_lds(g0, g1, z, z, 0);
#endif
}

// A fragment: 16x32 bf16, row-major source As[.][lda].
// lane l: M = l&15, K = kb..kb+7 and 16+kb..16+kb+7, kb = 8*(l>>4).
__device__ __forceinline__ v16bf load_a(const unsigned short* As, int lda,
                                        int mrow, int k0, int lane) {
    int m  = lane & 15;
    int kb = (lane >> 4) << 3;
    const unsigned short* p = As + (mrow + m) * lda + k0 + kb;
    Frag f;
    f.q[0] = *(const uint4v*)(p);
    f.q[1] = *(const uint4v*)(p + 16);
    return f.v;
}

// B fragment: 32x16 bf16 from transposed source BT[n][ldb] (n = output col).
// lane l: N = l&15, K = kb..kb+7 and 16+kb..16+kb+7, kb = 8*(l>>4).
__device__ __forceinline__ v16bf load_b(const unsigned short* BTs, int ldb,
                                        int ncol, int k0, int lane) {
    int n  = lane & 15;
    int kb = (lane >> 4) << 3;
    const unsigned short* p = BTs + (ncol + n) * ldb + k0 + kb;
    Frag f;
    f.q[0] = *(const uint4v*)(p);
    f.q[1] = *(const uint4v*)(p + 16);
    return f.v;
}

// C/D tile -> LDS bf16. lane l: col = l&15; VGPR j: row = 8*(l>>4)+j.
__device__ __forceinline__ void store_c_lds(unsigned short* Cs, int ldc,
                                            int mrow, int ncol, int lane, v8f c) {
    int col   = lane & 15;
    int rbase = mrow + ((lane >> 4) << 3);
#pragma unroll
    for (int j = 0; j < 8; ++j)
        Cs[(rbase + j) * ldc + ncol + col] = f2bf(c[j]);
}

// ---------------------------------------------------------------------------
// Kernel 0: fp32 [K][N] -> bf16 transposed [N][K]
// ---------------------------------------------------------------------------
extern "C" __global__ void wconv_kernel(const float* __restrict__ src,
                                        unsigned short* __restrict__ dst,
                                        int K, int N) {
    int idx = blockIdx.x * blockDim.x + threadIdx.x;
    if (idx >= K * N) return;
    int n = idx / K;
    int k = idx - n * K;
    dst[idx] = f2bf(src[(size_t)k * N + n]);
}

// ---------------------------------------------------------------------------
// Main fused flash-SVD FFN kernel. One block = 64 token rows, 8 waves.
// LDS (dynamic, 104 KB): PQ 32K | Vs 32K | Hs 8K | Us 32K.
// ---------------------------------------------------------------------------
extern "C" __global__ __launch_bounds__(256)
void flashsvd_ffn_kernel(const float* __restrict__ x,
                         const unsigned short* __restrict__ U1T,  // [RANK][D_MODEL]
                         const unsigned short* __restrict__ V1T,  // [D_FF][RANK]
                         const float* __restrict__ b1,
                         const unsigned short* __restrict__ U2T,  // [RANK][D_FF]
                         const unsigned short* __restrict__ V2T,  // [D_MODEL][RANK]
                         const float* __restrict__ b2,
                         float* __restrict__ out) {
    extern __shared__ __align__(16) unsigned short smem[];
    unsigned short* PQ  = smem;                 // [64][256] bf16: P then Q
    unsigned short* Vs  = smem + 64 * 256;      // [64][256]  V1/V2 chunk
    unsigned short* Hs  = Vs  + 64 * 256;       // [64][64]   H chunk
    unsigned short* Us  = Hs  + 64 * 64;        // [256][64]  U2 chunk
    unsigned short* Xs  = Vs;                   // stage-1 alias: [64][32]
    unsigned short* U1s = Vs + 64 * 32;         // stage-1 alias: [256][32]

    const unsigned ldsBase = __builtin_amdgcn_groupstaticsize();  // dynamic LDS origin
    const unsigned OFF_VS  = ldsBase + 64 * 256 * 2;              // byte offsets for TDM
    const unsigned OFF_U1S = OFF_VS + 64 * 32 * 2;
    const unsigned OFF_US  = ldsBase + (64 * 256 + 64 * 256 + 64 * 64) * 2;

    const int tid  = threadIdx.x;
    const int lane = tid & 31;
    const int w    = tid >> 5;                  // wave 0..7
    const int m0   = blockIdx.x * M_TILE;

    const int rt  = w >> 1;                     // row tile 0..3 (16 rows each)
    const int nb0 = (w & 1) * 128;              // 8 col tiles over RANK=256
    const int cb2 = (w & 1) * 2;                // 2 col tiles over a 64-wide chunk

    // ---------------- Phase 1: P[64,256] = X[64,1024] @ U1 -----------------
    v8f accP[8] = {};
    for (int k0 = 0; k0 < D_MODEL; k0 += 32) {
        if (w == 0) {   // U1T chunk [256 rows][64 B], row stride 2048 B -> TDM
            tdm_load_2d(OFF_U1S, U1T + k0, 64, 256, 2 * D_MODEL);
            __builtin_amdgcn_s_wait_tensorcnt(0);
        }
        {   // X chunk [64][32] fp32 -> bf16 LDS (needs convert: manual path)
            int row = tid >> 2, cs = (tid & 3) * 8;
            const float* src = x + (size_t)(m0 + row) * D_MODEL + k0 + cs;
            float4v f0 = *(const float4v*)(src);
            float4v f1 = *(const float4v*)(src + 4);
            union { unsigned short u[8]; uint4v q; } t;
            t.u[0] = f2bf(f0.x); t.u[1] = f2bf(f0.y);
            t.u[2] = f2bf(f0.z); t.u[3] = f2bf(f0.w);
            t.u[4] = f2bf(f1.x); t.u[5] = f2bf(f1.y);
            t.u[6] = f2bf(f1.z); t.u[7] = f2bf(f1.w);
            *(uint4v*)(Xs + row * 32 + cs) = t.q;
        }
        __syncthreads();
        v16bf a = load_a(Xs, 32, rt * 16, 0, lane);
#pragma unroll
        for (int ct = 0; ct < 8; ++ct) {
            v16bf b = load_b(U1s, 32, nb0 + ct * 16, 0, lane);
            accP[ct] = WMMA_BF16(a, b, accP[ct]);
        }
        __syncthreads();
    }
#pragma unroll
    for (int ct = 0; ct < 8; ++ct)
        store_c_lds(PQ, RANK, rt * 16, nb0 + ct * 16, lane, accP[ct]);
    __syncthreads();

    // ------- Phase 2: stream f-chunks; H stays in LDS (flash tiling) -------
    v8f accQ[8] = {};
    for (int f0 = 0; f0 < D_FF; f0 += FC) {
        if (w == 0) {
            // V1T rows f0..f0+63 full width: contiguous 32 KB -> 1-row TDM
            tdm_load_2d(OFF_VS, V1T + (size_t)f0 * RANK, 32768, 1, 32768);
            // U2T cols f0..f0+63 (256 rows x 128 B, stride 8192 B) -> 2D TDM
            tdm_load_2d(OFF_US, U2T + f0, 128, 256, 2 * D_FF);
            __builtin_amdgcn_s_wait_tensorcnt(0);
        }
        __syncthreads();

        // 2a: Hc[64,64] = gelu(P @ V1c + b1); wave -> (rt, cols (cb2+j)*16)
        v8f h[2] = {};
        for (int k0 = 0; k0 < RANK; k0 += 32) {
            v16bf a = load_a(PQ, RANK, rt * 16, k0, lane);
#pragma unroll
            for (int j = 0; j < 2; ++j) {
                v16bf b = load_b(Vs, RANK, (cb2 + j) * 16, k0, lane);
                h[j] = WMMA_BF16(a, b, h[j]);
            }
        }
#pragma unroll
        for (int j = 0; j < 2; ++j) {
            int ncol  = (cb2 + j) * 16;
            int col   = lane & 15;
            float bias = b1[f0 + ncol + col];
            int rbase = rt * 16 + ((lane >> 4) << 3);
#pragma unroll
            for (int i = 0; i < 8; ++i) {
                float v = h[j][i] + bias;
                Hs[(rbase + i) * FC + ncol + col] = f2bf(gelu_tanh(v));
            }
        }
        __syncthreads();

        // 2b: Q[64,256] += Hc @ U2c (K = 64)
        for (int k0 = 0; k0 < FC; k0 += 32) {
            v16bf a = load_a(Hs, FC, rt * 16, k0, lane);
#pragma unroll
            for (int ct = 0; ct < 8; ++ct) {
                v16bf b = load_b(Us, FC, nb0 + ct * 16, k0, lane);
                accQ[ct] = WMMA_BF16(a, b, accQ[ct]);
            }
        }
        __syncthreads();
    }

    // ---------------- Phase 3: out[64,1024] = Q @ V2 + b2 ------------------
#pragma unroll
    for (int ct = 0; ct < 8; ++ct)
        store_c_lds(PQ, RANK, rt * 16, nb0 + ct * 16, lane, accQ[ct]);
    __syncthreads();

    for (int n0 = 0; n0 < D_MODEL; n0 += 64) {
        if (w == 0) {   // V2T rows n0..n0+63: contiguous 32 KB -> TDM
            tdm_load_2d(OFF_VS, V2T + (size_t)n0 * RANK, 32768, 1, 32768);
            __builtin_amdgcn_s_wait_tensorcnt(0);
        }
        __syncthreads();

        v8f o[2] = {};
        for (int k0 = 0; k0 < RANK; k0 += 32) {
            v16bf a = load_a(PQ, RANK, rt * 16, k0, lane);
#pragma unroll
            for (int j = 0; j < 2; ++j) {
                v16bf b = load_b(Vs, RANK, (cb2 + j) * 16, k0, lane);
                o[j] = WMMA_BF16(a, b, o[j]);
            }
        }
#pragma unroll
        for (int j = 0; j < 2; ++j) {
            int ncol  = n0 + (cb2 + j) * 16 + (lane & 15);
            float bias = b2[ncol];
            int rbase = m0 + rt * 16 + ((lane >> 4) << 3);
#pragma unroll
            for (int i = 0; i < 8; ++i)
                out[(size_t)(rbase + i) * D_MODEL + ncol] = o[j][i] + bias;
        }
        __syncthreads();
    }
}

// ---------------------------------------------------------------------------
extern "C" void kernel_launch(void* const* d_in, const int* in_sizes, int n_in,
                              void* d_out, int out_size, void* d_ws, size_t ws_size,
                              hipStream_t stream) {
    (void)in_sizes; (void)n_in; (void)out_size; (void)ws_size;
    const float* x  = (const float*)d_in[0];
    const float* U1 = (const float*)d_in[1];
    const float* V1 = (const float*)d_in[2];
    const float* b1 = (const float*)d_in[3];
    const float* U2 = (const float*)d_in[4];
    const float* V2 = (const float*)d_in[5];
    const float* b2 = (const float*)d_in[6];
    float* out = (float*)d_out;

    unsigned short* U1T = (unsigned short*)d_ws;            // [256][1024]
    unsigned short* V1T = U1T + 256 * 1024;                 // [4096][256]
    unsigned short* U2T = V1T + 4096 * 256;                 // [256][4096]
    unsigned short* V2T = U2T + 256 * 4096;                 // [1024][256]

    dim3 blk(256);
    wconv_kernel<<<dim3((1024 * 256 + 255) / 256), blk, 0, stream>>>(U1, U1T, 1024, 256);
    wconv_kernel<<<dim3((256 * 4096 + 255) / 256), blk, 0, stream>>>(V1, V1T, 256, 4096);
    wconv_kernel<<<dim3((4096 * 256 + 255) / 256), blk, 0, stream>>>(U2, U2T, 4096, 256);
    wconv_kernel<<<dim3((256 * 1024 + 255) / 256), blk, 0, stream>>>(V2, V2T, 256, 1024);

    size_t smem = (size_t)(64 * 256 + 64 * 256 + 64 * 64 + 256 * 64) * sizeof(unsigned short);
    flashsvd_ffn_kernel<<<dim3(NROWS / M_TILE), blk, smem, stream>>>(
        x, U1T, V1T, b1, U2T, V2T, b2, out);
}
